// SelfAttention_me_5059471475066
// MI455X (gfx1250) — matrix operand order repaired and checked by reading, not compiled, not run on hardware
//
#include <hip/hip_runtime.h>
#include <math.h>

// ---------------------------------------------------------------------------
// Self-attention (B=8, S=2048, D=1024), fp32 I/O, bf16 WMMA internally.
//   V = x@W1^T+b1 ; K = x@W2^T+b2 ; Q = V
//   S = (K @ Q^T) / sqrt(D) ; P = softmax_rows(S) ; out = P @ V
// Pipeline: cvt->bf16, proj GEMMs (WMMA), score GEMM (WMMA), softmax, PV GEMM.
// ---------------------------------------------------------------------------

typedef __bf16 bf16_t;
typedef bf16_t v16bf __attribute__((ext_vector_type(16)));
typedef bf16_t v8bf  __attribute__((ext_vector_type(8)));
typedef float  v8f   __attribute__((ext_vector_type(8)));

#define DIM   1024
#define SEQ   2048
#define NBATCH 8
#define MTOT  (NBATCH * SEQ)   // 16384

static __device__ __forceinline__ bf16_t f2bf(float f) {
    unsigned u = __builtin_bit_cast(unsigned, f);
    unsigned r = u + 0x7FFFu + ((u >> 16) & 1u);   // round-to-nearest-even
    unsigned short h = (unsigned short)(r >> 16);
    return __builtin_bit_cast(bf16_t, h);
}
static __device__ __forceinline__ float bf2f(bf16_t b) {
    unsigned short h = __builtin_bit_cast(unsigned short, b);
    unsigned u = ((unsigned)h) << 16;
    return __builtin_bit_cast(float, u);
}

// ------------------------- fp32 -> bf16 conversion -------------------------
__global__ void cvt_f32_bf16_kernel(const float* __restrict__ src,
                                    bf16_t* __restrict__ dst, long n) {
    long i = (long)blockIdx.x * blockDim.x + threadIdx.x;
    long stride = (long)gridDim.x * blockDim.x;
    for (; i < n; i += stride) dst[i] = f2bf(src[i]);
}

// ----------------------------- WMMA NT GEMM --------------------------------
// C[m,n] = scale * sum_k A[m,k]*B[n,k] + bias[n]
// A: [M,K] bf16 row-major (lda), B: [N,K] bf16 row-major (ldb).
// Block = 256 threads = 8 waves (4 in M x 2 in N); block tile 128x128;
// wave tile 32x64 = 2x4 fragments of 16x16, K stepped by 32.
union Frag16 { v16bf v; v8bf h[2]; };

__global__ __launch_bounds__(256)
void gemm_nt_wmma_kernel(const bf16_t* __restrict__ A, long bsA,
                         const bf16_t* __restrict__ B, long bsB,
                         const float* __restrict__ bias,
                         bf16_t* __restrict__ Cb,       // bf16 out (or null)
                         float*  __restrict__ Cf,       // fp32 out (or null)
                         long bsC, int ldc,
                         bf16_t* __restrict__ Ct,       // transposed bf16 out (or null)
                         int tRows, long tBatchStride, int tLd,
                         int K, int lda, int ldb,
                         float scale) {
    const int lane = threadIdx.x & 31;
    const int wave = threadIdx.x >> 5;
    const int wm = wave & 3;            // 0..3 -> M
    const int wn = wave >> 2;           // 0..1 -> N
    const int mWave = blockIdx.y * 128 + wm * 32;
    const int nWave = blockIdx.x * 128 + wn * 64;

    const bf16_t* Ab = A + (long)blockIdx.z * bsA;
    const bf16_t* Bb = B + (long)blockIdx.z * bsB;

    const int laneLo = lane & 15;
    const int kA = (lane >> 4) << 3;    // 0 or 8   (A frag k-sub-offset)
    const int kB = (lane >> 4) << 3;    // 0 or 8   (B frag k-sub-offset)

    v8f acc[2][4] = {};

    for (int k = 0; k < K; k += 32) {
        Frag16 afr[2], bfr[4];
        #pragma unroll
        for (int i = 0; i < 2; ++i) {
            const bf16_t* pa = Ab + (long)(mWave + i * 16 + laneLo) * lda + k + kA;
            afr[i].h[0] = *(const v8bf*)(pa);
            afr[i].h[1] = *(const v8bf*)(pa + 16);
        }
        #pragma unroll
        for (int j = 0; j < 4; ++j) {
            const bf16_t* pb = Bb + (long)(nWave + j * 16 + laneLo) * ldb + k + kB;
            bfr[j].h[0] = *(const v8bf*)(pb);
            bfr[j].h[1] = *(const v8bf*)(pb + 16);
        }
        #pragma unroll
        for (int i = 0; i < 2; ++i)
            #pragma unroll
            for (int j = 0; j < 4; ++j)
                acc[i][j] = __builtin_amdgcn_wmma_f32_16x16x32_bf16(
                    false, afr[i].v, false, bfr[j].v,
                    (short)0, acc[i][j], false, false);
    }

    // Epilogue. C/D fragment layout: lane<16 -> M=v, lane>=16 -> M=v+8; N=lane&15.
    const int mLane = (lane >> 4) << 3;
    #pragma unroll
    for (int i = 0; i < 2; ++i) {
        const int m0 = mWave + i * 16 + mLane;
        #pragma unroll
        for (int j = 0; j < 4; ++j) {
            const int n = nWave + j * 16 + laneLo;
            const float bv = bias ? bias[n] : 0.0f;
            float vals[8];
            #pragma unroll
            for (int v = 0; v < 8; ++v) vals[v] = acc[i][j][v] * scale + bv;

            if (Cf) {
                float* cp = Cf + (long)blockIdx.z * bsC + (long)m0 * ldc + n;
                #pragma unroll
                for (int v = 0; v < 8; ++v) cp[(long)v * ldc] = vals[v];
            }
            if (Cb) {
                bf16_t* cp = Cb + (long)blockIdx.z * bsC + (long)m0 * ldc + n;
                #pragma unroll
                for (int v = 0; v < 8; ++v) cp[(long)v * ldc] = f2bf(vals[v]);
            }
            if (Ct) {   // Ct[b][n][m%tRows], 8 contiguous m's -> one b128 store
                const int bb = m0 / tRows;
                const int r  = m0 % tRows;
                v8bf tv;
                #pragma unroll
                for (int v = 0; v < 8; ++v) tv[v] = f2bf(vals[v]);
                *(v8bf*)(Ct + (long)bb * tBatchStride + (long)n * tLd + r) = tv;
            }
        }
    }
}

// ------------------------- row softmax (in place) --------------------------
// One 256-thread block per row of 2048 bf16 values.
__global__ __launch_bounds__(256)
void softmax_rows_kernel(bf16_t* __restrict__ S) {
    __shared__ float red[256];
    const int t = threadIdx.x;
    bf16_t* p = S + (long)blockIdx.x * SEQ;

    float x[8];
    float mx = -INFINITY;
    #pragma unroll
    for (int i = 0; i < 8; ++i) {
        x[i] = bf2f(p[t + i * 256]);
        mx = fmaxf(mx, x[i]);
    }
    red[t] = mx; __syncthreads();
    for (int s = 128; s > 0; s >>= 1) {
        if (t < s) red[t] = fmaxf(red[t], red[t + s]);
        __syncthreads();
    }
    mx = red[0]; __syncthreads();

    float sum = 0.0f;
    #pragma unroll
    for (int i = 0; i < 8; ++i) {
        x[i] = __expf(x[i] - mx);
        sum += x[i];
    }
    red[t] = sum; __syncthreads();
    for (int s = 128; s > 0; s >>= 1) {
        if (t < s) red[t] += red[t + s];
        __syncthreads();
    }
    const float inv = 1.0f / red[0];
    #pragma unroll
    for (int i = 0; i < 8; ++i) p[t + i * 256] = f2bf(x[i] * inv);
}

// ------------------------------- launcher ----------------------------------
extern "C" void kernel_launch(void* const* d_in, const int* in_sizes, int n_in,
                              void* d_out, int out_size, void* d_ws, size_t ws_size,
                              hipStream_t stream) {
    const float* x  = (const float*)d_in[0];
    const float* W1 = (const float*)d_in[1];
    const float* b1 = (const float*)d_in[2];
    const float* W2 = (const float*)d_in[3];
    const float* b2 = (const float*)d_in[4];
    float* out = (float*)d_out;

    char* ws = (char*)d_ws;
    size_t off = 0;
    bf16_t* xbf = (bf16_t*)(ws + off); off += (size_t)MTOT * DIM * 2;        // 32 MB
    bf16_t* Vbf = (bf16_t*)(ws + off); off += (size_t)MTOT * DIM * 2;        // 32 MB
    bf16_t* Rbf = (bf16_t*)(ws + off); off += (size_t)MTOT * DIM * 2;        // 32 MB
    bf16_t* Vt  = (bf16_t*)(ws + off); off += (size_t)NBATCH * DIM * SEQ * 2;// 32 MB
    bf16_t* W1b = (bf16_t*)(ws + off); off += (size_t)DIM * DIM * 2;         //  2 MB
    bf16_t* W2b = (bf16_t*)(ws + off); off += (size_t)DIM * DIM * 2;         //  2 MB
    bf16_t* Sbf = (bf16_t*)(ws + off); off += (size_t)MTOT * SEQ * 2;        // 64 MB

    const float scale = 1.0f / sqrtf((float)DIM);   // 1/32

    // 1) fp32 -> bf16
    cvt_f32_bf16_kernel<<<4096, 256, 0, stream>>>(x,  xbf, (long)MTOT * DIM);
    cvt_f32_bf16_kernel<<<512,  256, 0, stream>>>(W1, W1b, (long)DIM * DIM);
    cvt_f32_bf16_kernel<<<512,  256, 0, stream>>>(W2, W2b, (long)DIM * DIM);

    // 2) V = x@W1^T + b1  (bf16 out, plus transposed copy Vt[b][d][s])
    {
        dim3 grid(DIM / 128, MTOT / 128, 1);
        gemm_nt_wmma_kernel<<<grid, 256, 0, stream>>>(
            xbf, 0, W1b, 0, b1,
            Vbf, nullptr, 0, DIM,
            Vt, SEQ, (long)DIM * SEQ, SEQ,
            DIM, DIM, DIM, 1.0f);
    }
    // 3) R (= K projection) = x@W2^T + b2
    {
        dim3 grid(DIM / 128, MTOT / 128, 1);
        gemm_nt_wmma_kernel<<<grid, 256, 0, stream>>>(
            xbf, 0, W2b, 0, b2,
            Rbf, nullptr, 0, DIM,
            nullptr, 1, 0, 1,
            DIM, DIM, DIM, 1.0f);
    }
    // 4) scores S[b,r,q] = scale * R[b,r,:] . V[b,q,:]   (bf16 out)
    {
        dim3 grid(SEQ / 128, SEQ / 128, NBATCH);
        gemm_nt_wmma_kernel<<<grid, 256, 0, stream>>>(
            Rbf, (long)SEQ * DIM, Vbf, (long)SEQ * DIM, nullptr,
            Sbf, nullptr, (long)SEQ * SEQ, SEQ,
            nullptr, 1, 0, 1,
            DIM, DIM, DIM, scale);
    }
    // 5) row softmax in place
    softmax_rows_kernel<<<MTOT, 256, 0, stream>>>(Sbf);

    // 6) out[b,r,d] = sum_q P[b,r,q] * V[b,q,d]   (A = P, B = Vt[b][d][q], fp32 out)
    {
        dim3 grid(DIM / 128, SEQ / 128, NBATCH);
        gemm_nt_wmma_kernel<<<grid, 256, 0, stream>>>(
            Sbf, (long)SEQ * SEQ, Vt, (long)DIM * SEQ, nullptr,
            nullptr, out, (long)SEQ * DIM, DIM,
            nullptr, 1, 0, 1,
            SEQ, SEQ, SEQ, 1.0f);
    }
}
